// CNNEmbeddings_41592463294983
// MI455X (gfx1250) — matrix operand order repaired and checked by reading, not compiled, not run on hardware
//
#include <hip/hip_runtime.h>
#include <hip/hip_bf16.h>

typedef __bf16 bf16_t;
typedef __attribute__((ext_vector_type(16))) __bf16 v16bf;
typedef __attribute__((ext_vector_type(8)))  float  v8f;

#define L_SEQ      2048
#define NBATCH     32
#define NCH        768
#define KR         35     // real reduction length: 7 taps * 5 classes
#define LDS_STRIDE 772    // 768 + 4 pad (dwords): 8*772 % 64 != 0 -> no bank conflict

// ---------------------------------------------------------------------------
// Kernel 0: build the GEMM B-matrix (768 x 64, k-padded) as bf16 hi/lo WMMA
// fragments in d_ws. Contiguous-k layout (same as A) for V_WMMA_F32_16X16X32_BF16 B
// operand: lane n (0-15) holds column ct*16+n, k = kslab*32 + e (e=0..15);
// lanes 16-31 hold same columns, k = kslab*32 + 16 + e.
// frags[ct][slab][hilo][lane][e], each fragment = 32 lanes * 16 bf16 = 512 el.
// ---------------------------------------------------------------------------
__global__ void build_bfrag(const float* __restrict__ W3,
                            const float* __restrict__ W5,
                            const float* __restrict__ W7,
                            bf16_t* __restrict__ frags) {
  int bid  = blockIdx.x;         // 0..95 : (ct, slab)
  int ct   = bid >> 1;           // channel tile 0..47
  int s    = bid & 1;            // k-slab 0..1
  int lane = threadIdx.x;        // 0..31
  int c     = ct * 16 + (lane & 15);
  int kbase = s * 32 + ((lane >= 16) ? 16 : 0);

  bf16_t hi_v[16], lo_v[16];
#pragma unroll
  for (int e = 0; e < 16; ++e) {
    int j = kbase + e;           // reduction index = tap*5 + class
    float w = 0.0f;
    if (j < KR) {
      int o = j / 5;             // tap 0..6  (seq offset = o - 3)
      int i = j - o * 5;         // input class 0..4
      if (c < 256) {             // W3: offsets -1..+1 -> o in [2,4]
        if (o >= 2 && o <= 4) w = W3[(c * 5 + i) * 3 + (o - 2)];
      } else if (c < 512) {      // W5: offsets -2..+2 -> o in [1,5]
        if (o >= 1 && o <= 5) w = W5[((c - 256) * 5 + i) * 5 + (o - 1)];
      } else {                   // W7: all 7 taps
        w = W7[((c - 512) * 5 + i) * 7 + o];
      }
    }
    bf16_t h  = (bf16_t)w;             // RNE high part
    float rem = w - (float)h;          // residual
    hi_v[e] = h;
    lo_v[e] = (bf16_t)rem;
  }
  // hi fragment id = (ct*2+s)*2, lo fragment = +1; 512 bf16 per fragment
  size_t base = (size_t)((ct * 2 + s) * 2) * 512 + (size_t)lane * 16;
#pragma unroll
  for (int e = 0; e < 16; ++e) {
    frags[base + e]       = hi_v[e];
    frags[base + 512 + e] = lo_v[e];
  }
}

// ---------------------------------------------------------------------------
// Kernel 1: fused one-hot-conv-as-GEMM (bf16 WMMA, hi/lo split) + mask +
// LayerNorm. One 256-thread (8 wave32) workgroup per 16 sequence positions.
// ---------------------------------------------------------------------------
__global__ void __launch_bounds__(256)
cnn_ln_kernel(const int* __restrict__ ids,
              const float* __restrict__ amask,
              const bf16_t* __restrict__ frags,
              const float* __restrict__ gamma,
              const float* __restrict__ beta,
              float* __restrict__ out) {
  __shared__ float h_lds[16 * LDS_STRIDE];   // 16 positions x 768 ch (padded)
  __shared__ float psum[256];
  __shared__ float psq[256];

  int wg   = blockIdx.x;         // 0..4095
  int b    = wg >> 7;            // batch
  int pt   = wg & 127;           // position tile
  int pos0 = pt * 16;

  int tid  = threadIdx.x;
  int wave = tid >> 5;           // 0..7
  int lane = tid & 31;
  int m    = lane & 15;          // A-matrix row (position within tile)
  int p    = pos0 + m;
  bool hiHalf = lane >= 16;

  // Gather the 7 tap ids for this row, packed 3 bits each (7 = OOB sentinel).
  unsigned packed = 0;
#pragma unroll
  for (int o = 0; o < 7; ++o) {
    int q = p + o - 3;
    unsigned idv = 7u;
    if (q >= 0 && q < L_SEQ) idv = ((unsigned)ids[b * L_SEQ + q]) & 7u;
    packed |= idv << (3 * o);
  }

  // Build A fragments (exact 0/1 in bf16) in contiguous k order (same as B):
  // lanes 0-15: k in {0..15}; lanes 16-31: k in {16..31}.
  v16bf a0, a1;
#pragma unroll
  for (int e = 0; e < 16; ++e) {
    int k = (hiHalf ? 16 : 0) + e;
    float v0 = 0.0f, v1 = 0.0f;
    {
      int j = k;                 // slab 0
      if (j < KR) {
        int o = j / 5, i = j - o * 5;
        v0 = (((packed >> (3 * o)) & 7u) == (unsigned)i) ? 1.0f : 0.0f;
      }
    }
    {
      int j = k + 32;            // slab 1
      if (j < KR) {
        int o = j / 5, i = j - o * 5;
        v1 = (((packed >> (3 * o)) & 7u) == (unsigned)i) ? 1.0f : 0.0f;
      }
    }
    a0[e] = (bf16_t)v0;
    a1[e] = (bf16_t)v1;
  }

  // Each wave owns 6 channel tiles: ct = wave*6 + t, 48 tiles of 16 channels.
  const v16bf* f = (const v16bf*)frags;  // 32 v16bf per fragment
#pragma unroll
  for (int t = 0; t < 6; ++t) {
    int ct = wave * 6 + t;
    v16bf bh0 = f[(size_t)((ct * 2 + 0) * 2 + 0) * 32 + lane];
    v16bf bl0 = f[(size_t)((ct * 2 + 0) * 2 + 1) * 32 + lane];
    v16bf bh1 = f[(size_t)((ct * 2 + 1) * 2 + 0) * 32 + lane];
    v16bf bl1 = f[(size_t)((ct * 2 + 1) * 2 + 1) * 32 + lane];
    v8f acc = {};
    acc = __builtin_amdgcn_wmma_f32_16x16x32_bf16(false, a0, false, bh0,
                                                  (short)0, acc, false, false);
    acc = __builtin_amdgcn_wmma_f32_16x16x32_bf16(false, a0, false, bl0,
                                                  (short)0, acc, false, false);
    acc = __builtin_amdgcn_wmma_f32_16x16x32_bf16(false, a1, false, bh1,
                                                  (short)0, acc, false, false);
    acc = __builtin_amdgcn_wmma_f32_16x16x32_bf16(false, a1, false, bl1,
                                                  (short)0, acc, false, false);
    // D layout: lane n=lane&15 is the column; VGPR r -> row r (+8 for lanes>=16)
    int n  = lane & 15;
    int mb = hiHalf ? 8 : 0;
#pragma unroll
    for (int r = 0; r < 8; ++r)
      h_lds[(mb + r) * LDS_STRIDE + ct * 16 + n] = acc[r];
  }
  __syncthreads();

  // Masked LayerNorm over 768 channels: 16 threads per position, 48 ch each.
  int pp  = tid >> 4;            // position 0..15
  int seg = tid & 15;            // channel segment
  float msk = amask[b * L_SEQ + pos0 + pp];
  int c0 = seg * 48;
  float s = 0.0f, s2 = 0.0f;
#pragma unroll 4
  for (int c = c0; c < c0 + 48; ++c) {
    float v = h_lds[pp * LDS_STRIDE + c] * msk;
    s  += v;
    s2 += v * v;
  }
  psum[pp * 16 + seg] = s;
  psq [pp * 16 + seg] = s2;
  __syncthreads();

  float S = 0.0f, S2 = 0.0f;
#pragma unroll
  for (int q = 0; q < 16; ++q) {
    S  += psum[pp * 16 + q];
    S2 += psq [pp * 16 + q];
  }
  float mu   = S * (1.0f / NCH);
  float var  = S2 * (1.0f / NCH) - mu * mu;
  float rstd = rsqrtf(var + 1e-12f);

  size_t ob = ((size_t)(b * L_SEQ + pos0 + pp)) * NCH;
#pragma unroll 4
  for (int c = c0; c < c0 + 48; ++c) {
    float v = h_lds[pp * LDS_STRIDE + c] * msk;
    out[ob + c] = (v - mu) * rstd * gamma[c] + beta[c];
  }
}

// ---------------------------------------------------------------------------
// Launch: inputs in setup_inputs() order.
// ---------------------------------------------------------------------------
extern "C" void kernel_launch(void* const* d_in, const int* in_sizes, int n_in,
                              void* d_out, int out_size, void* d_ws, size_t ws_size,
                              hipStream_t stream) {
  const int*   ids   = (const int*)  d_in[0];   // input_ids [32,2048]
  const float* amask = (const float*)d_in[1];   // attention_mask [32,2048]
  const float* W3    = (const float*)d_in[2];   // [256,5,3]
  const float* W5    = (const float*)d_in[3];   // [256,5,5]
  const float* W7    = (const float*)d_in[4];   // [256,5,7]
  const float* gamma = (const float*)d_in[5];   // [768]
  const float* beta  = (const float*)d_in[6];   // [768]
  float* outp = (float*)d_out;                  // [32,2048,768] f32
  bf16_t* frags = (bf16_t*)d_ws;                // 48*2*2*512*2B = 196608 B

  // Recomputed every call (deterministic; graph-capture safe).
  build_bfrag<<<96, 32, 0, stream>>>(W3, W5, W7, frags);
  cnn_ln_kernel<<<NBATCH * (L_SEQ / 16), 256, 0, stream>>>(
      ids, amask, frags, gamma, beta, outp);
}
